// WeightedDilatedWindowAttention_28209345200726
// MI455X (gfx1250) — matrix operand order repaired and checked by reading, not compiled, not run on hardware
//
#include <hip/hip_runtime.h>
#include <stdint.h>

typedef __attribute__((ext_vector_type(16))) _Float16 v16h;
typedef __attribute__((ext_vector_type(8)))  float    v8f;

#define H_IMG 192
#define W_IMG 192
#define WSZ   12
#define ST    6
#define NHEAD 8
#define DHEAD 24

// LDS layout in dwords (total 7168 dwords = 28 KB)
#define V_OFF  0      // V:     64 rows x 16 dwords (32 halves, d 24..31 zero)
#define S_OFF  1024   // S:     64x64 f32
#define Q_OFF  5120   // Q:     64 rows x 16 dwords
#define K_OFF  6144   // K:     64 rows x 16 dwords
#define P_OFF  5120   // P:     64 rows x 32 dwords (reuses Q+K region)
#define LDS_DW 7168

struct ABreg { union { uint32_t u[8]; _Float16 h[16]; v16h v; }; };

__global__ __launch_bounds__(32)
void wdwa_attn_kernel(const float* __restrict__ x,
                      const float* __restrict__ fc,   // freqs_cis (8,8,2) [cos,sin]
                      float* __restrict__ out)
{
    __shared__ uint32_t lds[LDS_DW];
    const int lane = threadIdx.x;
    const int pid  = blockIdx.x;              // 0..8191
    const int head = pid & 7;
    const int widx = pid >> 3;                // window index (ih*16 + iw)*4 + g
    const int g    = widx & 3;
    const int iw   = (widx >> 2) & 15;
    const int ih   = widx >> 6;
    const int sh_r = (g & 1) ? ST : 0;
    const int sh_c = (g & 2) ? ST : 0;

    _Float16* qh = (_Float16*)(lds + Q_OFF);
    _Float16* kh = (_Float16*)(lds + K_OFF);
    _Float16* vh = (_Float16*)(lds + V_OFF);
    float*    sf = (float*)   (lds + S_OFF);
    _Float16* ph = (_Float16*)(lds + P_OFF);

    const int cstr = H_IMG * W_IMG;
    const int cq = head * DHEAD;          // q channel base
    const int ck = 192 + head * DHEAD;    // k channel base
    const int cv = 384 + head * DHEAD;    // v channel base

    // ---- Stage Q,K (with hs_pe rotation) and V into LDS as f16 ----
    #pragma unroll
    for (int rr = 0; rr < 2; ++rr) {
        const int l  = lane + rr * 32;            // seq position 0..63
        const int hy = l >> 3, wx = l & 7;
        int tr = ih * WSZ + hy + sh_r;
        int tc = iw * WSZ + wx + sh_c;
        int rs = (tr < H_IMG) ? tr : (2 * H_IMG - 2 - tr);   // reflect pad
        int cs = (tc < W_IMG) ? tc : (2 * W_IMG - 2 - tc);
        const float* xp = x + (size_t)rs * W_IMG + cs;

        #pragma unroll
        for (int cp = 0; cp < 8; ++cp) {
            const float chh = fc[(hy * 8 + cp) * 2 + 0];
            const float shh = fc[(hy * 8 + cp) * 2 + 1];
            const float cww = fc[(wx * 8 + cp) * 2 + 0];
            const float sww = fc[(wx * 8 + cp) * 2 + 1];
            // q: rotate (a,b) by theta[hy], then (b,c) by theta[wx]
            {
                float a = xp[(size_t)(cq + 3 * cp + 0) * cstr];
                float b = xp[(size_t)(cq + 3 * cp + 1) * cstr];
                float c = xp[(size_t)(cq + 3 * cp + 2) * cstr];
                float a2 = a * chh - b * shh;
                float b2 = a * shh + b * chh;
                qh[l * 32 + 3 * cp + 0] = (_Float16)a2;
                qh[l * 32 + 3 * cp + 1] = (_Float16)(b2 * cww - c * sww);
                qh[l * 32 + 3 * cp + 2] = (_Float16)(b2 * sww + c * cww);
            }
            // k: same rotation
            {
                float a = xp[(size_t)(ck + 3 * cp + 0) * cstr];
                float b = xp[(size_t)(ck + 3 * cp + 1) * cstr];
                float c = xp[(size_t)(ck + 3 * cp + 2) * cstr];
                float a2 = a * chh - b * shh;
                float b2 = a * shh + b * chh;
                kh[l * 32 + 3 * cp + 0] = (_Float16)a2;
                kh[l * 32 + 3 * cp + 1] = (_Float16)(b2 * cww - c * sww);
                kh[l * 32 + 3 * cp + 2] = (_Float16)(b2 * sww + c * cww);
            }
            // v: raw
            {
                vh[l * 32 + 3 * cp + 0] = (_Float16)xp[(size_t)(cv + 3 * cp + 0) * cstr];
                vh[l * 32 + 3 * cp + 1] = (_Float16)xp[(size_t)(cv + 3 * cp + 1) * cstr];
                vh[l * 32 + 3 * cp + 2] = (_Float16)xp[(size_t)(cv + 3 * cp + 2) * cstr];
            }
        }
        #pragma unroll
        for (int dp = 24; dp < 32; ++dp) {      // zero K-pad
            qh[l * 32 + dp] = (_Float16)0.0f;
            kh[l * 32 + dp] = (_Float16)0.0f;
            vh[l * 32 + dp] = (_Float16)0.0f;
        }
    }
    __syncthreads();   // single-wave WG: lowers to waits/NOP, keeps ordering explicit

    const uint32_t* qd = lds + Q_OFF;
    const uint32_t* kd = lds + K_OFF;
    const int lr = lane & 15;
    const int hi = lane >> 4;

    // ---- S = Q * K^T / sqrt(24): 4x4 tiles of v_wmma_f32_16x16x32_f16 ----
    #pragma unroll
    for (int mt = 0; mt < 4; ++mt) {
        ABreg A;
        {   // A 16x32 f16, this code's contiguous k order: lanes<16 rows, K 0..15; lanes>=16 K 16..31
            int r = mt * 16 + lr;
            int kb = hi * 8;
            #pragma unroll
            for (int i2 = 0; i2 < 8; ++i2)
                A.u[i2] = qd[r * 16 + kb + i2];
        }
        #pragma unroll
        for (int nt = 0; nt < 4; ++nt) {
            ABreg B;
            {   // B 32x16, same contiguous k order as A: lanes<16 col n, K 0..15; lanes>=16 K 16..31
                int n = nt * 16 + lr;
                int kb = hi * 8;
                #pragma unroll
                for (int i2 = 0; i2 < 8; ++i2)
                    B.u[i2] = kd[n * 16 + kb + i2];
            }
            v8f acc = {};
            acc = __builtin_amdgcn_wmma_f32_16x16x32_f16(
                    false, A.v, false, B.v, (short)0, acc, false, false);
            const int col   = nt * 16 + lr;
            const int rbase = mt * 16 + hi * 8;
            #pragma unroll
            for (int i2 = 0; i2 < 8; ++i2)
                sf[(rbase + i2) * 64 + col] = acc[i2] * 0.20412414523193154f; // 1/sqrt(24)
        }
    }
    __syncthreads();

    // ---- Softmax over keys; write P (f16) into reused Q/K region ----
    #pragma unroll
    for (int rr = 0; rr < 2; ++rr) {
        const int l = lane + rr * 32;
        float mx = -1e30f;
        for (int m = 0; m < 64; ++m) mx = fmaxf(mx, sf[l * 64 + m]);
        float sum = 0.0f;
        for (int m = 0; m < 64; ++m) sum += __expf(sf[l * 64 + m] - mx);
        const float inv = 1.0f / sum;
        for (int m = 0; m < 64; ++m)
            ph[l * 64 + m] = (_Float16)(__expf(sf[l * 64 + m] - mx) * inv);
    }
    __syncthreads();

    // ---- O = P (64x64) * V (64x24, padded 32): 4x2 tiles, 2 K-steps each ----
    const uint32_t* pd = lds + P_OFF;   // row stride 32 dwords
    #pragma unroll
    for (int mt = 0; mt < 4; ++mt) {
        ABreg A0, A1;
        #pragma unroll
        for (int kt = 0; kt < 2; ++kt) {
            ABreg* Ak = kt ? &A1 : &A0;
            int r  = mt * 16 + lr;
            int kb = hi * 8 + kt * 16;
            #pragma unroll
            for (int i2 = 0; i2 < 8; ++i2)
                Ak->u[i2] = pd[r * 32 + kb + i2];
        }
        #pragma unroll
        for (int nt = 0; nt < 2; ++nt) {
            v8f acc = {};
            #pragma unroll
            for (int kt = 0; kt < 2; ++kt) {
                ABreg B;
                const int n  = nt * 16 + lr;
                const int k0 = kt * 32 + hi * 16;
                #pragma unroll
                for (int i2 = 0; i2 < 8; ++i2) {
                    B.h[2 * i2 + 0] = vh[(k0 + 2 * i2 + 0) * 32 + n];
                    B.h[2 * i2 + 1] = vh[(k0 + 2 * i2 + 1) * 32 + n];
                }
                acc = __builtin_amdgcn_wmma_f32_16x16x32_f16(
                        false, kt ? A1.v : A0.v, false, B.v, (short)0, acc, false, false);
            }
            // ---- Fused reverse(): overlap-add into output image via atomics ----
            const int d = nt * 16 + lr;
            if (d < DHEAD) {
                const int rbase = mt * 16 + hi * 8;
                const int cg = head * DHEAD + d;   // output channel (0..191)
                #pragma unroll
                for (int i2 = 0; i2 < 8; ++i2) {
                    const int l  = rbase + i2;
                    const int hy = l >> 3, wx = l & 7;
                    const int pr = ih * WSZ + hy + sh_r;
                    const int pc = iw * WSZ + wx + sh_c;
                    if (pr < H_IMG && pc < W_IMG)
                        atomicAdd(out + ((size_t)cg * H_IMG + pr) * W_IMG + pc, acc[i2]);
                }
            }
        }
    }
}

__global__ void wdwa_zero_kernel(float* __restrict__ out, int n) {
    int i = blockIdx.x * blockDim.x + threadIdx.x;
    if (i < n) out[i] = 0.0f;
}

__global__ void wdwa_div_kernel(float* __restrict__ out,
                                const float* __restrict__ w, int n) {
    int i = blockIdx.x * blockDim.x + threadIdx.x;
    if (i < n) out[i] = out[i] / w[i % (H_IMG * W_IMG)];
}

extern "C" void kernel_launch(void* const* d_in, const int* in_sizes, int n_in,
                              void* d_out, int out_size, void* d_ws, size_t ws_size,
                              hipStream_t stream) {
    const float* x  = (const float*)d_in[0];   // (1,576,192,192)
    const float* fc = (const float*)d_in[1];   // (8,8,2)
    const float* w  = (const float*)d_in[2];   // (1,1,192,192)
    float* out = (float*)d_out;                // (1,192,192,192)

    const int n = 192 * H_IMG * W_IMG;
    wdwa_zero_kernel<<<(n + 255) / 256, 256, 0, stream>>>(out, n);
    wdwa_attn_kernel<<<1024 * NHEAD, 32, 0, stream>>>(x, fc, out);
    wdwa_div_kernel<<<(n + 255) / 256, 256, 0, stream>>>(out, w, n);
}
